// main_model_86569360818469
// MI455X (gfx1250) — matrix operand order repaired and checked by reading, not compiled, not run on hardware
//
#include <hip/hip_runtime.h>
#include <hip/hip_bf16.h>
#include <stdint.h>

typedef _Float16 v16h __attribute__((ext_vector_type(16)));
typedef _Float16 v8h  __attribute__((ext_vector_type(8)));
typedef float    v8f  __attribute__((ext_vector_type(8)));

#define NWIN   4096      // 2 * (16/2) * (112/7) * (112/7)
#define NTOK   98        // 2*7*7 tokens per window
#define TOKP   112       // padded to 7 WMMA M-tiles
#define CDIM   96
#define C3     288
#define KPAD   128       // attn K padded to 4 chunks of 32
#define ATTN_SCALE 0.17677669529663687f  // 32^-0.5

// ---- LDS layout (bytes), total 117760 ----
#define XW_OFF   0        // 112x96  f16 = 21504  ; reused later as attention output (AO)
#define QK_OFF   21504    // 112x192 f16 = 43008  ; reused later as proj output (PO 112x96)
#define VT_OFF   64512    // 96x128  f16 = 24576  (v transposed, zero-padded cols)
#define SP_OFF   89088    // 112x128 f16 = 28672  (scores / probs, in-place softmax)
#define XRAW_OFF 64512    // transient gather buffer f32 98x96 = 37632 (aliases VT+part of SP)
#define SMEM_BYTES 117760

__device__ __forceinline__ v16h ld2(const _Float16* p, int step) {
  v8h lo = *(const v8h*)p;
  v8h hi = *(const v8h*)(p + step);
  return __builtin_shufflevector(lo, hi, 0,1,2,3,4,5,6,7,8,9,10,11,12,13,14,15);
}
__device__ __forceinline__ v8f wmma16(v16h a, v16h b, v8f c) {
  return __builtin_amdgcn_wmma_f32_16x16x32_f16(false, a, false, b, (short)0, c, false, false);
}
__device__ __forceinline__ float wred_sum(float v) {
#pragma unroll
  for (int o = 16; o > 0; o >>= 1) v += __shfl_xor(v, o, 32);
  return v;
}
__device__ __forceinline__ float wred_max(float v) {
#pragma unroll
  for (int o = 16; o > 0; o >>= 1) v = fmaxf(v, __shfl_xor(v, o, 32));
  return v;
}

__global__ void cvt_f16_kernel(const float* __restrict__ s, _Float16* __restrict__ d, int n) {
  int i = blockIdx.x * 256 + threadIdx.x;
  if (i < n) d[i] = (_Float16)s[i];
}

__global__ __launch_bounds__(256)
void swin_window_kernel(const float* __restrict__ x,
                        const float* __restrict__ nw, const float* __restrict__ nb,
                        const _Float16* __restrict__ wqkv, const float* __restrict__ qkv_b,
                        const float* __restrict__ bias_tbl, const float* __restrict__ amask,
                        const _Float16* __restrict__ wproj, const float* __restrict__ proj_b,
                        const _Float16* __restrict__ wfc1, const float* __restrict__ fc1_b,
                        float* __restrict__ out) {
  extern __shared__ __align__(16) char smem[];
  _Float16* XW = (_Float16*)(smem + XW_OFF);
  _Float16* AO = XW;                        // reuse after QKV
  _Float16* QK = (_Float16*)(smem + QK_OFF);
  _Float16* PO = QK;                        // reuse after attention
  _Float16* VT = (_Float16*)(smem + VT_OFF);
  _Float16* SP = (_Float16*)(smem + SP_OFF);
  float*    XR = (float*)(smem + XRAW_OFF);

  const int tid  = threadIdx.x;
  const int lane = tid & 31;
  // wave index is uniform within a wave -> force into SGPR so tile loops,
  // bounds and branches become scalar (s_cbranch, SALU address math)
  const int wave = __builtin_amdgcn_readfirstlane(tid >> 5);
  const int lr   = lane & 15;
  const int hip_ = (lane >> 4) & 1;         // 0: lanes 0-15, 1: lanes 16-31

  const int win = blockIdx.x;
  const int b_  = win >> 11;
  const int rem = win & 2047;
  const int wd  = rem >> 8, wh = (rem >> 4) & 15, ww = rem & 15;

  // ---------- Phase A: async gather of shifted window (f32) + weight prefetch ----------
  {
    const char* wp0 = (const char*)wqkv;
    for (int i = tid; i < 432; i += 256) __builtin_prefetch(wp0 + i * 128, 0, 2);
    const char* wp1 = (const char*)wproj;
    const char* wp2 = (const char*)wfc1;
    for (int i = tid; i < 144; i += 256) { __builtin_prefetch(wp1 + i * 128, 0, 2);
                                           __builtin_prefetch(wp2 + i * 128, 0, 2); }
  }
  for (int i = tid; i < NTOK * 24; i += 256) {       // 98 rows x 24 b128 chunks
    int t = i / 24, j = i % 24;
    int md = t / 49, tr = t % 49, mh = tr / 7, mw = tr % 7;
    int dp = (wd * 2 + md + 1) & 15;
    int hp = wh * 7 + mh + 3; if (hp >= 112) hp -= 112;
    int wp = ww * 7 + mw + 3; if (wp >= 112) wp -= 112;
    size_t grow = (((size_t)b_ * 16 + dp) * 112 + hp) * 112 + wp;
    uint64_t ga = (uint64_t)(const char*)(x + grow * CDIM) + (uint64_t)j * 16;
    uint32_t lo = (uint32_t)(size_t)(smem + XRAW_OFF + (size_t)(t * 24 + j) * 16);
    asm volatile("global_load_async_to_lds_b128 %0, %1, off" :: "v"(lo), "v"(ga) : "memory");
  }
  // zero XW pad rows 98..111 (disjoint from XR region)
  for (int i = tid; i < 672; i += 256) ((uint32_t*)(XW + NTOK * CDIM))[i] = 0u;
  asm volatile("s_wait_asynccnt 0x0" ::: "memory");
  __syncthreads();

  // ---------- LayerNorm (wave per token) into f16 XW ----------
  {
    float w0 = nw[lane], w1 = nw[lane + 32], w2 = nw[lane + 64];
    float b0 = nb[lane], b1 = nb[lane + 32], b2 = nb[lane + 64];
    for (int t = wave; t < NTOK; t += 8) {
      const float* r = XR + t * CDIM;
      float v0 = r[lane], v1 = r[lane + 32], v2 = r[lane + 64];
      float s  = wred_sum(v0 + v1 + v2);
      float s2 = wred_sum(v0 * v0 + v1 * v1 + v2 * v2);
      float mu = s * (1.0f / 96.0f);
      float inv = rsqrtf(s2 * (1.0f / 96.0f) - mu * mu + 1e-5f);
      _Float16* dst = XW + t * CDIM;
      dst[lane]      = (_Float16)((v0 - mu) * inv * w0 + b0);
      dst[lane + 32] = (_Float16)((v1 - mu) * inv * w1 + b1);
      dst[lane + 64] = (_Float16)((v2 - mu) * inv * w2 + b2);
    }
  }
  __syncthreads();
  // zero VT (overwrites dead XR)
  for (int i = tid; i < (CDIM * KPAD) / 2; i += 256) ((uint32_t*)VT)[i] = 0u;
  __syncthreads();

  // ---------- Phase B1: Q,K = XW(112x96) @ Wqkv[0:192]^T, q scaled ----------
  for (int tile = wave; tile < 84; tile += 8) {      // 7 M-tiles x 12 N-tiles
    int mt = tile / 12, nt = tile % 12;              // nt<6: q, 6..11: k
    v8f c = {};
    int arow = mt * 16 + lr;
    int akb  = hip_ ? 16 : 0;
    int ncol = nt * 16 + lr;
    const _Float16* bp = wqkv + ncol * CDIM + (hip_ ? 16 : 0);
#pragma unroll
    for (int kc = 0; kc < 3; ++kc) {
      v16h a = ld2(XW + arow * CDIM + kc * 32 + akb, 8);
      v16h b = ld2(bp + kc * 32, 8);
      c = wmma16(a, b, c);
    }
    float mult  = (nt < 6) ? ATTN_SCALE : 1.0f;      // nt scalar -> s_cselect
    float biasm = qkv_b[ncol] * mult;
    _Float16* dst = QK + (mt * 16 + hip_ * 8) * 192 + ncol;
#pragma unroll
    for (int r = 0; r < 8; ++r)
      dst[r * 192] = (_Float16)(c[r] * mult + biasm);
  }
  // ---------- Phase B2: V^T tiles (contiguous b128 stores) ----------
  for (int tile = wave; tile < 42; tile += 8) {      // 7 M-tiles x 6 N-tiles
    int mt = tile / 6, nt = tile % 6;
    v8f c = {};
    int arow = mt * 16 + lr;
    int akb  = hip_ ? 16 : 0;
    int ncol = 192 + nt * 16 + lr;
    const _Float16* bp = wqkv + ncol * CDIM + (hip_ ? 16 : 0);
#pragma unroll
    for (int kc = 0; kc < 3; ++kc) {
      v16h a = ld2(XW + arow * CDIM + kc * 32 + akb, 8);
      v16h b = ld2(bp + kc * 32, 8);
      c = wmma16(a, b, c);
    }
    float bias = qkv_b[ncol];
    _Float16* dst = VT + (ncol - 192) * KPAD + mt * 16 + hip_ * 8;
#pragma unroll
    for (int r = 0; r < 8; ++r)
      dst[r] = (_Float16)(c[r] + bias);
  }
  __syncthreads();

  // ---------- Phase C: per-head attention ----------
  for (int h = 0; h < 3; ++h) {
    // scores S = q @ k^T (+bias +mask), K=32
    for (int tile = wave; tile < 49; tile += 8) {    // 7x7 tiles
      int mt = tile / 7, nt = tile % 7;
      v8f c = {};
      int qrow = mt * 16 + lr;
      int kcol = nt * 16 + lr;
      v16h a = ld2(QK + qrow * 192 + h * 32 + (hip_ ? 16 : 0), 8);
      v16h b = ld2(QK + kcol * 192 + 96 + h * 32 + (hip_ ? 16 : 0), 8);
      c = wmma16(a, b, c);

      int m = nt * 16 + lr;                          // per-lane col (hoisted decomposition)
      int md2 = m / 49, mr = m % 49, mh2 = mr / 7, mw2 = mr % 7;
      int mrel = -md2 * 169 - mh2 * 13 - mw2 + 1 * 169 + 6 * 13 + 6;
      const float* maskcol = amask + (size_t)(win & 2047) * (NTOK * NTOK) + m;
      int nbase = mt * 16 + hip_ * 8;
      _Float16* spc = SP + m;
      if (mt < 6 && nt < 6) {                        // scalar condition: interior tiles
#pragma unroll
        for (int r = 0; r < 8; ++r) {
          int n = nbase + r;
          int nd = n / 49, nr = n % 49, nh = nr / 7, nwv = nr % 7;
          int idx = nd * 169 + nh * 13 + nwv + mrel;
          float s = c[r] + bias_tbl[idx * 3 + h] + maskcol[n * NTOK];
          spc[n * KPAD] = (_Float16)s;
        }
      } else {
        bool mok = m < NTOK;
#pragma unroll
        for (int r = 0; r < 8; ++r) {
          int n = nbase + r;
          float s = c[r];
          if (mok && n < NTOK) {
            int nd = n / 49, nr = n % 49, nh = nr / 7, nwv = nr % 7;
            int idx = nd * 169 + nh * 13 + nwv + mrel;
            s += bias_tbl[idx * 3 + h] + maskcol[n * NTOK];
          }
          spc[n * KPAD] = (_Float16)s;
        }
      }
    }
    __syncthreads();
    // softmax rows (in place), pad cols -> 0
    for (int row = wave; row < TOKP; row += 8) {
      _Float16* rp = SP + row * KPAD;
      float v0 = (lane      < NTOK) ? (float)rp[lane]      : -3e38f;
      float v1 = (lane + 32 < NTOK) ? (float)rp[lane + 32] : -3e38f;
      float v2 = (lane + 64 < NTOK) ? (float)rp[lane + 64] : -3e38f;
      float v3 = (lane + 96 < NTOK) ? (float)rp[lane + 96] : -3e38f;
      float mx = wred_max(fmaxf(fmaxf(v0, v1), fmaxf(v2, v3)));
      float e0 = (lane      < NTOK) ? __expf(v0 - mx) : 0.0f;
      float e1 = (lane + 32 < NTOK) ? __expf(v1 - mx) : 0.0f;
      float e2 = (lane + 64 < NTOK) ? __expf(v2 - mx) : 0.0f;
      float e3 = (lane + 96 < NTOK) ? __expf(v3 - mx) : 0.0f;
      float inv = 1.0f / wred_sum(e0 + e1 + e2 + e3);
      rp[lane]      = (_Float16)(e0 * inv);
      rp[lane + 32] = (_Float16)(e1 * inv);
      rp[lane + 64] = (_Float16)(e2 * inv);
      rp[lane + 96] = (_Float16)(e3 * inv);
    }
    __syncthreads();
    // O_h = P(112x128) @ VT_h  (VT row-major [d][m] serves as B), K=128
    for (int tile = wave; tile < 14; tile += 8) {    // 7 M-tiles x 2 N-tiles(16)
      int mt = tile / 2, ntv = tile % 2;
      v8f c = {};
      int prow = mt * 16 + lr;
      int dcol = h * 32 + ntv * 16 + lr;
#pragma unroll
      for (int kc = 0; kc < 4; ++kc) {
        v16h a = ld2(SP + prow * KPAD + kc * 32 + (hip_ ? 16 : 0), 8);
        v16h b = ld2(VT + dcol * KPAD + kc * 32 + (hip_ ? 16 : 0), 8);
        c = wmma16(a, b, c);
      }
      _Float16* dst = AO + (mt * 16 + hip_ * 8) * CDIM + dcol;
#pragma unroll
      for (int r = 0; r < 8; ++r)
        dst[r * CDIM] = (_Float16)c[r];
    }
    __syncthreads();
  }

  // ---------- Phase D: proj = AO(112x96) @ Wproj^T + b ----------
  for (int tile = wave; tile < 42; tile += 8) {      // 7 x 6
    int mt = tile / 6, nt = tile % 6;
    v8f c = {};
    int arow = mt * 16 + lr;
    int ncol = nt * 16 + lr;
    const _Float16* bp = wproj + ncol * CDIM + (hip_ ? 16 : 0);
#pragma unroll
    for (int kc = 0; kc < 3; ++kc) {
      v16h a = ld2(AO + arow * CDIM + kc * 32 + (hip_ ? 16 : 0), 8);
      v16h b = ld2(bp + kc * 32, 8);
      c = wmma16(a, b, c);
    }
    float bias = proj_b[ncol];
    _Float16* dst = PO + (mt * 16 + hip_ * 8) * CDIM + ncol;
#pragma unroll
    for (int r = 0; r < 8; ++r)
      dst[r * CDIM] = (_Float16)(c[r] + bias);
  }
  __syncthreads();

  // ---------- Phase E: fc1 + exact GELU + window-reverse scatter + residual ----------
  for (int tile = wave; tile < 42; tile += 8) {
    int mt = tile / 6, nt = tile % 6;
    v8f c = {};
    int arow = mt * 16 + lr;
    int ncol = nt * 16 + lr;
    const _Float16* bp = wfc1 + ncol * CDIM + (hip_ ? 16 : 0);
#pragma unroll
    for (int kc = 0; kc < 3; ++kc) {
      v16h a = ld2(PO + arow * CDIM + kc * 32 + (hip_ ? 16 : 0), 8);
      v16h b = ld2(bp + kc * 32, 8);
      c = wmma16(a, b, c);
    }
    float bias = fc1_b[ncol];
    int rmax = (mt < 6) ? 8 : 2;                     // scalar bound: pad rows only in last M-tile
#pragma unroll 2
    for (int r = 0; r < rmax; ++r) {
      int t = mt * 16 + r + hip_ * 8;
      if (t < NTOK) {
        float v = c[r] + bias;
        float g = 0.5f * v * (1.0f + erff(v * 0.70710678f));
        int md = t / 49, tr = t % 49, mh = tr / 7, mw = tr % 7;
        int dp = (wd * 2 + md + 1) & 15;
        int hp = wh * 7 + mh + 3; if (hp >= 112) hp -= 112;
        int wp = ww * 7 + mw + 3; if (wp >= 112) wp -= 112;
        size_t grow = (((size_t)b_ * 16 + dp) * 112 + hp) * 112 + wp;
        out[grow * CDIM + ncol] = x[grow * CDIM + ncol] + g;
      }
    }
  }
}

extern "C" void kernel_launch(void* const* d_in, const int* in_sizes, int n_in,
                              void* d_out, int out_size, void* d_ws, size_t ws_size,
                              hipStream_t stream) {
  (void)in_sizes; (void)n_in; (void)out_size; (void)ws_size;
  const float* x         = (const float*)d_in[0];
  const float* amask     = (const float*)d_in[1];
  const float* norm1_w   = (const float*)d_in[2];
  const float* norm1_b   = (const float*)d_in[3];
  const float* qkv_w     = (const float*)d_in[4];
  const float* qkv_b     = (const float*)d_in[5];
  const float* bias_tbl  = (const float*)d_in[6];
  const float* proj_w    = (const float*)d_in[7];
  const float* proj_b    = (const float*)d_in[8];
  const float* fc1_w     = (const float*)d_in[9];
  const float* fc1_b     = (const float*)d_in[10];
  float* out = (float*)d_out;

  char* ws = (char*)d_ws;
  _Float16* wqkvh  = (_Float16*)(ws);                      // 288*96 f16 = 55296 B
  _Float16* wprojh = (_Float16*)(ws + 55296);              // 96*96  f16 = 18432 B
  _Float16* wfc1h  = (_Float16*)(ws + 55296 + 18432);      // 96*96  f16 = 18432 B

  cvt_f16_kernel<<<(27648 + 255) / 256, 256, 0, stream>>>(qkv_w, wqkvh, 27648);
  cvt_f16_kernel<<<(9216 + 255) / 256, 256, 0, stream>>>(proj_w, wprojh, 9216);
  cvt_f16_kernel<<<(9216 + 255) / 256, 256, 0, stream>>>(fc1_w, wfc1h, 9216);

  swin_window_kernel<<<NWIN, 256, SMEM_BYTES, stream>>>(
      x, norm1_w, norm1_b, wqkvh, qkv_b, bias_tbl, amask,
      wprojh, proj_b, wfc1h, fc1_b, out);
}
